// FinalProjectModel_38087769981236
// MI455X (gfx1250) — compile-verified, matrix operand order checked by reading, not run on hardware
//
#include <hip/hip_runtime.h>
#include <hip/hip_bf16.h>
#include <math.h>

typedef __attribute__((ext_vector_type(16))) _Float16 v16h;
typedef __attribute__((ext_vector_type(8)))  _Float16 v8h;
typedef __attribute__((ext_vector_type(8)))  float    v8f;

static constexpr int BB = 8;     // batch
static constexpr int NN = 192;   // nodes
static constexpr int DD = 128;   // feature dim
static constexpr int KH = 32;    // SIG_HID

// ---------------------------------------------------------------------------
// WMMA tile loader: fills the 16-half A/B operand for v_wmma_f32_16x16x32_f16
// from a row-major fp16 matrix (rows row0..row0+15, K-chunk at k0..k0+31).
// ISA 7.12.2 layout: lane = hi*16 + m;  elements 0..7 -> K = k0 + hi*8 .. +7,
// elements 8..15 -> K = k0 + 16 + hi*8 .. +7.  Two aligned 16B loads per lane.
// ---------------------------------------------------------------------------
__device__ __forceinline__ v16h wmma_tile_rows(const _Float16* base, int row0,
                                               int ld, int k0, int lane) {
  int m = lane & 15, hi = lane >> 4;
  const _Float16* p = base + (size_t)(row0 + m) * ld + k0 + hi * 8;
  v8h lo = *(const v8h*)p;
  v8h hh = *(const v8h*)(p + 16);
  v16h r;
#pragma unroll
  for (int e = 0; e < 8; ++e) { r[e] = lo[e]; r[e + 8] = hh[e]; }
  return r;
}

// ---------------------------------------------------------------------------
// Stage A: h = ELU(LayerNorm(x @ W_fe^T + b_fe))   [B,N,4] -> [B,N,128]
// one block per (b,n) row, 128 threads (one per output channel)
// ---------------------------------------------------------------------------
__global__ void featln_kernel(const float* __restrict__ x,
                              const float* __restrict__ W_fe,
                              const float* __restrict__ b_fe,
                              const float* __restrict__ ln_g,
                              const float* __restrict__ ln_b,
                              float* __restrict__ h) {
  int bn = blockIdx.x;
  int d = threadIdx.x;
  const float* xr = x + (size_t)bn * 4;
  const float* w = W_fe + (size_t)d * 4;
  float v = fmaf(xr[0], w[0], fmaf(xr[1], w[1], fmaf(xr[2], w[2], xr[3] * w[3]))) + b_fe[d];

  __shared__ float red[DD];
  red[d] = v;
  __syncthreads();
  for (int s = DD / 2; s > 0; s >>= 1) {
    if (d < s) red[d] += red[d + s];
    __syncthreads();
  }
  float mu = red[0] * (1.0f / DD);
  __syncthreads();
  float dv = v - mu;
  red[d] = dv * dv;
  __syncthreads();
  for (int s = DD / 2; s > 0; s >>= 1) {
    if (d < s) red[d] += red[d + s];
    __syncthreads();
  }
  float var = red[0] * (1.0f / DD);
  v = dv * rsqrtf(var + 1e-5f) * ln_g[d] + ln_b[d];
  v = v > 0.0f ? v : expm1f(v);               // ELU
  h[(size_t)bn * DD + d] = v;
}

// ---------------------------------------------------------------------------
// Stage B: channel attention; scales h in place.  one block per batch.
// ---------------------------------------------------------------------------
__global__ void attn_kernel(const float* __restrict__ Wca1,
                            const float* __restrict__ bca1,
                            const float* __restrict__ Wca2,
                            const float* __restrict__ bca2,
                            float* __restrict__ h) {
  int b = blockIdx.x;
  int d = threadIdx.x;                        // 128
  float* hb = h + (size_t)b * NN * DD;
  float s = 0.0f;
  for (int n = 0; n < NN; ++n) s += hb[(size_t)n * DD + d];
  __shared__ float hm[DD];
  __shared__ float z1[16];
  hm[d] = s * (1.0f / NN);
  __syncthreads();
  if (d < 16) {
    float a = bca1[d];
    const float* w = Wca1 + (size_t)d * DD;
    for (int k = 0; k < DD; ++k) a = fmaf(hm[k], w[k], a);
    z1[d] = fmaxf(a, 0.0f);
  }
  __syncthreads();
  float a = bca2[d];
  const float* w2 = Wca2 + (size_t)d * 16;
  for (int k = 0; k < 16; ++k) a = fmaf(z1[k], w2[k], a);
  float attn = 1.0f / (1.0f + expf(-a));
  for (int n = 0; n < NN; ++n) hb[(size_t)n * DD + d] *= attn;
}

// ---------------------------------------------------------------------------
// DGF prep: fp16 copies of X (row-major + transposed), fp16 Wp, row norms.
// grid = B + 1 blocks x 256 threads (last block converts Wp)
// ---------------------------------------------------------------------------
__global__ void conv_kernel(const float* __restrict__ h,
                            const float* __restrict__ Wp,
                            _Float16* __restrict__ Xh,
                            _Float16* __restrict__ XhT,
                            float* __restrict__ rn,
                            _Float16* __restrict__ Wph) {
  int b = blockIdx.x;
  if (b == (int)gridDim.x - 1) {
    for (int idx = threadIdx.x; idx < DD * DD; idx += blockDim.x)
      Wph[idx] = (_Float16)Wp[idx];
    return;
  }
  const float* hb = h + (size_t)b * NN * DD;
  _Float16* xh = Xh + (size_t)b * NN * DD;
  _Float16* xt = XhT + (size_t)b * DD * NN;
  for (int idx = threadIdx.x; idx < NN * DD; idx += blockDim.x) {
    int i = idx / DD, d = idx % DD;
    _Float16 v = (_Float16)hb[idx];
    xh[idx] = v;
    xt[(size_t)d * NN + i] = v;
  }
  if (threadIdx.x < NN) {
    const float* row = hb + (size_t)threadIdx.x * DD;
    float s = 0.0f;
    for (int dd = 0; dd < DD; ++dd) {
      float v = (float)(_Float16)row[dd];     // match fp16-rounded Gram inputs
      s = fmaf(v, v, s);
    }
    rn[(size_t)b * NN + threadIdx.x] = s;
  }
}

// ---------------------------------------------------------------------------
// DGF head projections si/sj: [B,H,N,32], dots of depth 128 (VALU)
// grid = B*H blocks x 256 threads
// ---------------------------------------------------------------------------
__global__ void sig_kernel(const float* __restrict__ h,
                           const float* __restrict__ W1, int H,
                           float* __restrict__ si, float* __restrict__ sj) {
  int bh = blockIdx.x;
  int b = bh / H, hd = bh % H;
  const float* hb = h + (size_t)b * NN * DD;
  const float* Wrow = W1 + (size_t)hd * KH * (2 * DD);
  float* sib = si + ((size_t)b * H + hd) * NN * KH;
  float* sjb = sj + ((size_t)b * H + hd) * NN * KH;
  for (int idx = threadIdx.x; idx < NN * KH; idx += blockDim.x) {
    int n = idx >> 5, k = idx & 31;
    const float* xr = hb + (size_t)n * DD;
    const float* wr = Wrow + (size_t)k * (2 * DD);
    float sa = 0.0f, sb = 0.0f;
    for (int t = 0; t < DD; ++t) {
      float xv = xr[t];
      sa = fmaf(xv, wr[t], sa);
      sb = fmaf(xv, wr[DD + t], sb);
    }
    sib[idx] = sa;
    sjb[idx] = sb;
  }
}

// ---------------------------------------------------------------------------
// Pairwise distances via WMMA Gram matrix: dist = r_i + r_j - 2*(X X^T)_ij
// grid = dim3(12*12, B), block = 32 (one wave per 16x16 tile)
// ---------------------------------------------------------------------------
__global__ void gram_kernel(const _Float16* __restrict__ Xh,
                            const float* __restrict__ rn,
                            float* __restrict__ dist) {
  int b = blockIdx.y;
  int ti = blockIdx.x / 12, tj = blockIdx.x % 12;
  int lane = threadIdx.x;
  const _Float16* X = Xh + (size_t)b * NN * DD;
  v8f acc = {};
#pragma unroll
  for (int k0 = 0; k0 < DD; k0 += 32) {
    v16h a = wmma_tile_rows(X, ti * 16, DD, k0, lane);
    v16h bb = wmma_tile_rows(X, tj * 16, DD, k0, lane);
    acc = __builtin_amdgcn_wmma_f32_16x16x32_f16(false, a, false, bb,
                                                 (short)0, acc, false, false);
  }
  int n = lane & 15, hi = lane >> 4;
  const float* rb = rn + (size_t)b * NN;
  float* dd = dist + (size_t)b * NN * NN;
#pragma unroll
  for (int r = 0; r < 8; ++r) {
    int i = ti * 16 + r + 8 * hi;
    int j = tj * 16 + n;
    dd[(size_t)i * NN + j] = rb[i] + rb[j] - 2.0f * acc[r];
  }
}

// ---------------------------------------------------------------------------
// Adjacency: sigma = softplus(tanh(si+sj+b1)@w2 + b2); A = mean_h exp(-d/(2s^2+eps))
// grid = dim3(N, B), block = 192 (thread j)   -- the transcendental hot loop
// ---------------------------------------------------------------------------
__global__ void adj_kernel(const float* __restrict__ si,
                           const float* __restrict__ sj,
                           const float* __restrict__ b1,
                           const float* __restrict__ w2,
                           const float* __restrict__ b2,
                           const float* __restrict__ dist, int H,
                           _Float16* __restrict__ Ah) {
  int b = blockIdx.y, i = blockIdx.x;
  int j = threadIdx.x;                         // 192
  __shared__ float sil[4 * KH], b1l[4 * KH], w2l[4 * KH];
  for (int t = threadIdx.x; t < H * KH; t += blockDim.x) {
    int hh = t >> 5, k = t & 31;
    sil[t] = si[(((size_t)b * H + hh) * NN + i) * KH + k];
    b1l[t] = b1[t];
    w2l[t] = w2[t];
  }
  __syncthreads();
  float dv = dist[((size_t)b * NN + i) * NN + j];
  float acc = 0.0f;
  for (int hh = 0; hh < H; ++hh) {
    const float* sjr = sj + (((size_t)b * H + hh) * NN + j) * KH;
    float t = 0.0f;
#pragma unroll
    for (int k = 0; k < KH; ++k)
      t = fmaf(tanhf(sil[hh * KH + k] + sjr[k] + b1l[hh * KH + k]),
               w2l[hh * KH + k], t);
    t += b2[hh];
    float sp = t > 20.0f ? t : log1pf(expf(t));   // softplus
    acc += expf(-dv / (2.0f * sp * sp + 1e-6f));
  }
  Ah[((size_t)b * NN + i) * NN + j] = (_Float16)(acc / (float)H);
}

// ---------------------------------------------------------------------------
// hnew = A @ X  via WMMA (K = 192).  grid = dim3(12*8, B), block = 32.
// B-operand columns come from pre-transposed fp16 X (contiguous K loads).
// ---------------------------------------------------------------------------
__global__ void apply_kernel(const _Float16* __restrict__ Ah,
                             const _Float16* __restrict__ XhT,
                             _Float16* __restrict__ hnewh) {
  int b = blockIdx.y;
  int ti = blockIdx.x / 8, td = blockIdx.x % 8;
  int lane = threadIdx.x;
  const _Float16* A = Ah + (size_t)b * NN * NN;
  const _Float16* Bt = XhT + (size_t)b * DD * NN;
  v8f acc = {};
#pragma unroll
  for (int k0 = 0; k0 < NN; k0 += 32) {
    v16h a = wmma_tile_rows(A, ti * 16, NN, k0, lane);
    v16h bb = wmma_tile_rows(Bt, td * 16, NN, k0, lane);
    acc = __builtin_amdgcn_wmma_f32_16x16x32_f16(false, a, false, bb,
                                                 (short)0, acc, false, false);
  }
  int n = lane & 15, hi = lane >> 4;
  _Float16* out = hnewh + (size_t)b * NN * DD;
#pragma unroll
  for (int r = 0; r < 8; ++r) {
    int i = ti * 16 + r + 8 * hi;
    int d = td * 16 + n;
    out[(size_t)i * DD + d] = (_Float16)acc[r];
  }
}

// ---------------------------------------------------------------------------
// h = ELU(hnew @ Wp^T + bp + h)  via WMMA (K = 128).  grid = dim3(12*8, B).
// B-operand columns d are rows of Wp (contiguous).
// ---------------------------------------------------------------------------
__global__ void proj_kernel(const _Float16* __restrict__ hnewh,
                            const _Float16* __restrict__ Wph,
                            const float* __restrict__ bp,
                            float* __restrict__ h) {
  int b = blockIdx.y;
  int ti = blockIdx.x / 8, td = blockIdx.x % 8;
  int lane = threadIdx.x;
  const _Float16* A = hnewh + (size_t)b * NN * DD;
  v8f acc = {};
#pragma unroll
  for (int k0 = 0; k0 < DD; k0 += 32) {
    v16h a = wmma_tile_rows(A, ti * 16, DD, k0, lane);
    v16h bb = wmma_tile_rows(Wph, td * 16, DD, k0, lane);
    acc = __builtin_amdgcn_wmma_f32_16x16x32_f16(false, a, false, bb,
                                                 (short)0, acc, false, false);
  }
  int n = lane & 15, hi = lane >> 4;
  float* hb = h + (size_t)b * NN * DD;
#pragma unroll
  for (int r = 0; r < 8; ++r) {
    int i = ti * 16 + r + 8 * hi;
    int d = td * 16 + n;
    float v = acc[r] + bp[d] + hb[(size_t)i * DD + d];   // residual
    hb[(size_t)i * DD + d] = v > 0.0f ? v : expm1f(v);   // ELU
  }
}

// ---------------------------------------------------------------------------
// Head: feat = [mean_n h, max_n h]; z = leaky(bn(feat@Wc1^T+bc1)); out = z@Wc2^T+bc2
// grid = B blocks x 128 threads
// ---------------------------------------------------------------------------
__global__ void head_kernel(const float* __restrict__ h,
                            const float* __restrict__ Wc1,
                            const float* __restrict__ bc1,
                            const float* __restrict__ bn_g,
                            const float* __restrict__ bn_b,
                            const float* __restrict__ Wc2,
                            const float* __restrict__ bc2,
                            float* __restrict__ out) {
  int b = blockIdx.x;
  int d = threadIdx.x;                          // 128
  const float* hb = h + (size_t)b * NN * DD;
  float s = 0.0f, mx = -INFINITY;
  for (int n = 0; n < NN; ++n) {
    float v = hb[(size_t)n * DD + d];
    s += v;
    mx = fmaxf(mx, v);
  }
  __shared__ float feat[2 * DD];
  __shared__ float z1[64];
  feat[d] = s * (1.0f / NN);
  feat[DD + d] = mx;
  __syncthreads();
  if (d < 64) {
    float a = bc1[d];
    const float* w = Wc1 + (size_t)d * (2 * DD);
    for (int k = 0; k < 2 * DD; ++k) a = fmaf(feat[k], w[k], a);
    a = a * rsqrtf(1.0f + 1e-5f) * bn_g[d] + bn_b[d];
    z1[d] = a >= 0.0f ? a : 0.2f * a;
  }
  __syncthreads();
  if (d < 2) {
    float a = bc2[d];
    const float* w = Wc2 + (size_t)d * 64;
    for (int k = 0; k < 64; ++k) a = fmaf(z1[k], w[k], a);
    out[(size_t)b * 2 + d] = a;
  }
}

// ---------------------------------------------------------------------------
// Workspace layout (bytes, 256-aligned)
// ---------------------------------------------------------------------------
static constexpr size_t OFF_H     = 0;                       // f32 [B,N,D]
static constexpr size_t OFF_XH    = OFF_H     + 786432;      // f16 [B,N,D]
static constexpr size_t OFF_XHT   = OFF_XH    + 393216;      // f16 [B,D,N]
static constexpr size_t OFF_RN    = OFF_XHT   + 393216;      // f32 [B,N]
static constexpr size_t OFF_WPH   = OFF_RN    + 6144;        // f16 [D,D]
static constexpr size_t OFF_SI    = OFF_WPH   + 32768;       // f32 [B,4,N,32]
static constexpr size_t OFF_SJ    = OFF_SI    + 786432;      // f32 [B,4,N,32]
static constexpr size_t OFF_DIST  = OFF_SJ    + 786432;      // f32 [B,N,N]
static constexpr size_t OFF_AH    = OFF_DIST  + 1179648;     // f16 [B,N,N]
static constexpr size_t OFF_HNEW  = OFF_AH    + 589824;      // f16 [B,N,D]

extern "C" void kernel_launch(void* const* d_in, const int* in_sizes, int n_in,
                              void* d_out, int out_size, void* d_ws, size_t ws_size,
                              hipStream_t stream) {
  const float* x     = (const float*)d_in[0];
  const float* W_fe  = (const float*)d_in[1];
  const float* b_fe  = (const float*)d_in[2];
  const float* ln_g  = (const float*)d_in[3];
  const float* ln_b  = (const float*)d_in[4];
  const float* W_ca1 = (const float*)d_in[5];
  const float* b_ca1 = (const float*)d_in[6];
  const float* W_ca2 = (const float*)d_in[7];
  const float* b_ca2 = (const float*)d_in[8];
  const float* Wc1   = (const float*)d_in[21];
  const float* bc1   = (const float*)d_in[22];
  const float* bn_g  = (const float*)d_in[23];
  const float* bn_b  = (const float*)d_in[24];
  const float* Wc2   = (const float*)d_in[25];
  const float* bc2   = (const float*)d_in[26];

  char* ws = (char*)d_ws;
  float*    h     = (float*)(ws + OFF_H);
  _Float16* Xh    = (_Float16*)(ws + OFF_XH);
  _Float16* XhT   = (_Float16*)(ws + OFF_XHT);
  float*    rn    = (float*)(ws + OFF_RN);
  _Float16* Wph   = (_Float16*)(ws + OFF_WPH);
  float*    si    = (float*)(ws + OFF_SI);
  float*    sj    = (float*)(ws + OFF_SJ);
  float*    dist  = (float*)(ws + OFF_DIST);
  _Float16* Ah    = (_Float16*)(ws + OFF_AH);
  _Float16* hnewh = (_Float16*)(ws + OFF_HNEW);

  featln_kernel<<<BB * NN, DD, 0, stream>>>(x, W_fe, b_fe, ln_g, ln_b, h);
  attn_kernel<<<BB, DD, 0, stream>>>(W_ca1, b_ca1, W_ca2, b_ca2, h);

  for (int g = 0; g < 2; ++g) {
    int base = (g == 0) ? 9 : 15;
    int H    = (g == 0) ? 4 : 2;
    const float* W1 = (const float*)d_in[base + 0];
    const float* b1 = (const float*)d_in[base + 1];
    const float* w2 = (const float*)d_in[base + 2];
    const float* b2 = (const float*)d_in[base + 3];
    const float* Wp = (const float*)d_in[base + 4];
    const float* bp = (const float*)d_in[base + 5];

    conv_kernel<<<BB + 1, 256, 0, stream>>>(h, Wp, Xh, XhT, rn, Wph);
    sig_kernel<<<BB * H, 256, 0, stream>>>(h, W1, H, si, sj);
    gram_kernel<<<dim3(12 * 12, BB), 32, 0, stream>>>(Xh, rn, dist);
    adj_kernel<<<dim3(NN, BB), NN, 0, stream>>>(si, sj, b1, w2, b2, dist, H, Ah);
    apply_kernel<<<dim3(12 * 8, BB), 32, 0, stream>>>(Ah, XhT, hnewh);
    proj_kernel<<<dim3(12 * 8, BB), 32, 0, stream>>>(hnewh, Wph, bp, h);
  }

  head_kernel<<<BB, DD, 0, stream>>>(h, Wc1, bc1, bn_g, bn_b, Wc2, bc2,
                                     (float*)d_out);
}
